// ScaledQuantizedLinearINT4_52793738002495
// MI455X (gfx1250) — compile-verified, matrix operand order checked by reading, not run on hardware
//
#include <hip/hip_runtime.h>
#include <hip/hip_bf16.h>
#include <math.h>

// ---- problem dims (fixed by the reference) ----
#define BATCH 16384
#define IN_F  2048
#define OUT_F 2048

// ---- GEMM tiling ----
#define BM  128
#define BN  128
#define BK  64   // two 16x16x32 k-steps per staged tile
#define LDT 72   // LDS row stride in bf16 (144 B -> conflict-free b128 access)
#define NT  (IN_F / BK)   // 32 K-tiles

typedef __attribute__((ext_vector_type(16))) __bf16 v16bf;
typedef __attribute__((ext_vector_type(8)))  float  v8f;

// low 32 bits of a generic pointer to LDS == LDS byte offset (aperture is in high bits)
#define LDS_OFF(p) ((unsigned)(unsigned long long)(p))

// round-to-nearest-even f32 -> bf16 (bit pattern)
__device__ __forceinline__ unsigned short f32_to_bf16(float f) {
    unsigned int u = __float_as_uint(f);
    unsigned int r = u + 0x7FFFu + ((u >> 16) & 1u);
    return (unsigned short)(r >> 16);
}

// sign(v) * |v|^p  (v==0 -> 0, matching jnp.sign semantics)
__device__ __forceinline__ float spow(float v, float p) {
    float r = powf(fabsf(v), p);
    return v < 0.0f ? -r : r;
}

// ---------------------------------------------------------------------------
// Kernel 1: x_scaled = sign(x)*|x|^alpha, stored as bf16.  float4 -> 4x bf16.
// ---------------------------------------------------------------------------
__global__ __launch_bounds__(256) void scale_x_kernel(
    const float* __restrict__ x, const float* __restrict__ alpha,
    unsigned short* __restrict__ xb) {
    const float a = alpha[0];
    int idx = blockIdx.x * blockDim.x + threadIdx.x;   // one float4 per thread
    float4 v = reinterpret_cast<const float4*>(x)[idx];
    unsigned int s0 = f32_to_bf16(spow(v.x, a));
    unsigned int s1 = f32_to_bf16(spow(v.y, a));
    unsigned int s2 = f32_to_bf16(spow(v.z, a));
    unsigned int s3 = f32_to_bf16(spow(v.w, a));
    uint2 o;
    o.x = s0 | (s1 << 16);
    o.y = s2 | (s3 << 16);
    reinterpret_cast<uint2*>(xb)[idx] = o;
}

// ---------------------------------------------------------------------------
// Kernel 2: W_bf16[o,i] = bf16(centroids[W_q[o,i]]).  int4 -> 4x bf16.
// ---------------------------------------------------------------------------
__global__ __launch_bounds__(256) void dequant_w_kernel(
    const int* __restrict__ Wq, const float* __restrict__ cent,
    unsigned short* __restrict__ wb) {
    __shared__ unsigned short cb[16];
    if (threadIdx.x < 16) cb[threadIdx.x] = f32_to_bf16(cent[threadIdx.x]);
    __syncthreads();
    int idx = blockIdx.x * blockDim.x + threadIdx.x;   // one int4 per thread
    int4 q = reinterpret_cast<const int4*>(Wq)[idx];
    unsigned int s0 = cb[q.x & 15];
    unsigned int s1 = cb[q.y & 15];
    unsigned int s2 = cb[q.z & 15];
    unsigned int s3 = cb[q.w & 15];
    uint2 o;
    o.x = s0 | (s1 << 16);
    o.y = s2 | (s3 << 16);
    reinterpret_cast<uint2*>(wb)[idx] = o;
}

// ---------------------------------------------------------------------------
// Async staging of one 128x64 bf16 tile pair (A and B) into LDS.
// 1024 b128 chunks per matrix tile -> 4 async loads per thread per matrix.
// GVS addressing: saddr = uniform base, vaddr = per-lane 32-bit byte offset.
// ---------------------------------------------------------------------------
__device__ __forceinline__ void issue_tile_async(
    const unsigned short* __restrict__ xb, const unsigned short* __restrict__ wb,
    unsigned short* sAbuf, unsigned short* sBbuf,
    size_t arow0, size_t brow0, int kbase, int tid) {
#pragma unroll
    for (int j = 0; j < 4; ++j) {
        int linear = tid + j * 256;          // 0..1023
        int row    = linear >> 3;            // 8 b128 chunks per 64-half row
        int ch     = (linear & 7) * 8;       // bf16 offset within row
        unsigned ga = (unsigned)((((unsigned)arow0 + row) * IN_F + kbase + ch) * 2);
        unsigned la = LDS_OFF(&sAbuf[row * LDT + ch]);
        asm volatile("global_load_async_to_lds_b128 %0, %1, %2"
                     :: "v"(la), "v"(ga), "s"(xb) : "memory");
        unsigned gb = (unsigned)((((unsigned)brow0 + row) * IN_F + kbase + ch) * 2);
        unsigned lb = LDS_OFF(&sBbuf[row * LDT + ch]);
        asm volatile("global_load_async_to_lds_b128 %0, %1, %2"
                     :: "v"(lb), "v"(gb), "s"(wb) : "memory");
    }
}

// ---------------------------------------------------------------------------
// Kernel 3: bf16 WMMA GEMM  out[b,o] = spow( sum_i xb[b,i]*wb[o,i], 1/alpha )
//   256 threads = 8 wave32 waves; 128x128 tile; wave grid 2(M) x 4(N);
//   per wave 64x32 = 4x2 WMMA tiles; double-buffered async LDS pipeline.
// ---------------------------------------------------------------------------
__global__ __launch_bounds__(256) void gemm_bf16_kernel(
    const unsigned short* __restrict__ xb,   // [BATCH, IN_F] bf16
    const unsigned short* __restrict__ wb,   // [OUT_F, IN_F] bf16
    const float* __restrict__ alpha,
    float* __restrict__ out) {               // [BATCH, OUT_F] f32
    __shared__ unsigned short sA0[BM * LDT];
    __shared__ unsigned short sA1[BM * LDT];
    __shared__ unsigned short sB0[BN * LDT];
    __shared__ unsigned short sB1[BN * LDT];

    const int nblk = OUT_F / BN;             // 16
    const int bm = blockIdx.x / nblk;
    const int bn = blockIdx.x % nblk;
    const int tid  = threadIdx.x;
    const int lane = tid & 31;
    const int wid  = tid >> 5;
    const int wm = (wid & 1) * 64;           // wave M offset
    const int wn = (wid >> 1) * 32;          // wave N offset
    const int l16   = lane & 15;
    const int khalf = (lane >> 4) * 8;       // half-wave K sub-offset (A/B layout)

    const size_t arow0 = (size_t)bm * BM;
    const size_t brow0 = (size_t)bn * BN;

    v8f acc[4][2];
#pragma unroll
    for (int mi = 0; mi < 4; ++mi)
#pragma unroll
        for (int ni = 0; ni < 2; ++ni) {
            v8f z = {0.f, 0.f, 0.f, 0.f, 0.f, 0.f, 0.f, 0.f};
            acc[mi][ni] = z;
        }

    union FU { uint4 u[2]; v16bf v; };

    // Prologue: stage tile 0 into buffer 0.
    issue_tile_async(xb, wb, sA0, sB0, arow0, brow0, 0, tid);

    for (int t = 0; t < NT; ++t) {
        // Drain this wave's outstanding async loads (tile t), then sync so
        // every wave's tile-t data is visible and the "other" buffer is free.
        asm volatile("s_wait_asynccnt 0x0" ::: "memory");
        __syncthreads();

        unsigned short* curA = (t & 1) ? sA1 : sA0;
        unsigned short* curB = (t & 1) ? sB1 : sB0;
        if (t + 1 < NT) {
            unsigned short* nxtA = (t & 1) ? sA0 : sA1;
            unsigned short* nxtB = (t & 1) ? sB0 : sB1;
            issue_tile_async(xb, wb, nxtA, nxtB, arow0, brow0, (t + 1) * BK, tid);
        }

        // Two 16x16x32 k-steps per staged 64-wide tile.
#pragma unroll
        for (int ks = 0; ks < BK; ks += 32) {
            FU af[4], bfg[2];
#pragma unroll
            for (int mi = 0; mi < 4; ++mi) {
                int r = wm + mi * 16 + l16;
                af[mi].u[0] = *reinterpret_cast<const uint4*>(&curA[r * LDT + ks + khalf]);
                af[mi].u[1] = *reinterpret_cast<const uint4*>(&curA[r * LDT + ks + 16 + khalf]);
            }
#pragma unroll
            for (int ni = 0; ni < 2; ++ni) {
                int r = wn + ni * 16 + l16;
                bfg[ni].u[0] = *reinterpret_cast<const uint4*>(&curB[r * LDT + ks + khalf]);
                bfg[ni].u[1] = *reinterpret_cast<const uint4*>(&curB[r * LDT + ks + 16 + khalf]);
            }
#pragma unroll
            for (int mi = 0; mi < 4; ++mi)
#pragma unroll
                for (int ni = 0; ni < 2; ++ni)
                    acc[mi][ni] = __builtin_amdgcn_wmma_f32_16x16x32_bf16(
                        false, af[mi].v, false, bfg[ni].v,
                        (short)0, acc[mi][ni], false, false);
        }
    }

    // Epilogue: D layout = lane(0..15)->N, lane>=16 -> M+8; VGPR r -> M+r.
    const float inv_a = 1.0f / alpha[0];
    const int mhalf = (lane >> 4) * 8;
#pragma unroll
    for (int mi = 0; mi < 4; ++mi) {
#pragma unroll
        for (int ni = 0; ni < 2; ++ni) {
            size_t col = brow0 + wn + ni * 16 + l16;
#pragma unroll
            for (int r = 0; r < 8; ++r) {
                size_t row = arow0 + wm + mi * 16 + mhalf + r;
                out[row * OUT_F + col] = spow(acc[mi][ni][r], inv_a);
            }
        }
    }
}

// ---------------------------------------------------------------------------
extern "C" void kernel_launch(void* const* d_in, const int* in_sizes, int n_in,
                              void* d_out, int out_size, void* d_ws, size_t ws_size,
                              hipStream_t stream) {
    const float* x     = (const float*)d_in[0];   // [BATCH, IN_F]
    const float* cent  = (const float*)d_in[1];   // [16]
    const float* alpha = (const float*)d_in[2];   // [1]
    const int*   Wq    = (const int*)d_in[3];     // [OUT_F, IN_F]
    float* out = (float*)d_out;

    // workspace: bf16 x_scaled (67.1 MB) then bf16 W (8.4 MB) -- both L2-resident
    unsigned short* xb = (unsigned short*)d_ws;
    unsigned short* wb = xb + (size_t)BATCH * IN_F;

    scale_x_kernel<<<(BATCH * IN_F / 4) / 256, 256, 0, stream>>>(x, alpha, xb);
    dequant_w_kernel<<<(OUT_F * IN_F / 4) / 256, 256, 0, stream>>>(Wq, cent, wb);

    dim3 grid((BATCH / BM) * (OUT_F / BN));       // 128 * 16 = 2048 blocks
    gemm_bf16_kernel<<<grid, 256, 0, stream>>>(xb, wb, alpha, out);
}
